// GAT_14688788152986
// MI455X (gfx1250) — matrix operand order repaired and checked by reading, not compiled, not run on hardware
//
#include <hip/hip_runtime.h>
#include <hip/hip_bf16.h>
#include <math.h>

// GAT (2-layer) for MI455X / gfx1250. WMMA bf16 GEMMs + wave-per-edge scatter.
typedef __attribute__((ext_vector_type(16))) __bf16 v16bf;
typedef __attribute__((ext_vector_type(8)))  float  v8f;
typedef __attribute__((ext_vector_type(8)))  unsigned short us8;
typedef __attribute__((ext_vector_type(16))) unsigned short us16;

#define GAT_H   4
#define GAT_D   64
#define GAT_HD  256
#define NEG_SLOPE 0.2f

static __device__ __forceinline__ unsigned short f32_to_bf16_rne(float f) {
  unsigned int u = __float_as_uint(f);
  unsigned int r = u + 0x7FFFu + ((u >> 16) & 1u);
  return (unsigned short)(r >> 16);
}
static __device__ __forceinline__ void atomicMaxF(float* addr, float v) {
  if (v >= 0.0f) atomicMax((int*)addr, __float_as_int(v));
  else           atomicMin((unsigned int*)addr, __float_as_uint(v));
}
static __device__ __forceinline__ v16bf ld_frag16(const unsigned short* p) {
  return __builtin_bit_cast(v16bf, *(const us16*)p);   // 32B -> 2x global_load_b128
}

// ---------------- elementwise helpers ----------------
__global__ void k_cvt_bf16(const float* __restrict__ in, unsigned short* __restrict__ out, int count) {
  int i = blockIdx.x * blockDim.x + threadIdx.x;
  if (i < count) out[i] = f32_to_bf16_rne(in[i]);
}
// f32 [K x NC] row-major -> bf16 [NC x K] (transposed, so GEMM B-fragments are contiguous)
__global__ void k_cvt_bf16_t(const float* __restrict__ in, unsigned short* __restrict__ out,
                             int K, int NC) {
  int i = blockIdx.x * blockDim.x + threadIdx.x;
  if (i >= K * NC) return;
  int k = i / NC, c = i % NC;
  out[(size_t)c * K + k] = f32_to_bf16_rne(in[i]);
}
__global__ void k_fill(float* __restrict__ p, float v, int count) {
  int i = blockIdx.x * blockDim.x + threadIdx.x;
  if (i < count) p[i] = v;
}

// ---------------- WMMA bf16 GEMM: C[MxNC] = A[MxK] * Bt^T ----------------
// A row-major bf16 [M x K]; Bt col-major bf16 [NC x K]; C row-major f32.
// One wave computes a 16x64 output strip: one A fragment feeds 4 WMMAs per
// 32-deep k-step (v_wmma_f32_16x16x32_bf16). Compile-time K/NC so every
// fragment load is base + constant immediate and the k-loop fully unrolls.
template <int K, int NC>
__global__ void k_gemm_wmma_bf16(const unsigned short* __restrict__ A,
                                 const unsigned short* __restrict__ Bt,
                                 float* __restrict__ C, int M) {
  constexpr int stripsN = NC / 64;     // 64-wide column strips
  int wave = (int)((blockIdx.x * blockDim.x + threadIdx.x) >> 5);
  int lane = threadIdx.x & 31;
  int tilesM = (M + 15) >> 4;
  int tm = wave / stripsN;
  int tc = wave % stripsN;
  if (tm >= tilesM) return;            // wave-uniform exit: EXEC all-ones inside

  int g  = lane >> 4;                  // lane group (0/1)
  int mr = lane & 15;                  // A row within tile / B,C column within tile
  int row  = tm * 16 + mr;
  int rowc = row < M ? row : (M - 1);  // clamp loads; stores guarded below
  const unsigned short* Arow = A + (size_t)rowc * K;
  // lane's B column for tile j is (tc*64 + 16*j + mr); K-run starts at 16*g
  const unsigned short* Bcol = Bt + ((size_t)(tc * 64 + mr)) * K + 16 * g;

  v8f acc[4] = {{}, {}, {}, {}};
#pragma unroll
  for (int kb = 0; kb < K; kb += 32) {
    // A 16x32 fragment: lane holds K in [kb+16g, +16) -> 2 x b128
    us8 a0 = *(const us8*)(Arow + kb + 16 * g);
    us8 a1 = *(const us8*)(Arow + kb + 16 * g + 8);
    v16bf a = __builtin_bit_cast(
        v16bf, __builtin_shufflevector(a0, a1, 0, 1, 2, 3, 4, 5, 6, 7,
                                       8, 9, 10, 11, 12, 13, 14, 15));
    // Issue all 4 B fragments first (8 x b128 in a clause), then the 4 WMMAs,
    // so the scheduler can start WMMA j while B j+1.. are still in flight.
    v16bf b0 = ld_frag16(Bcol + 0 * 16 * K + kb);
    v16bf b1 = ld_frag16(Bcol + 1 * 16 * K + kb);
    v16bf b2 = ld_frag16(Bcol + 2 * 16 * K + kb);
    v16bf b3 = ld_frag16(Bcol + 3 * 16 * K + kb);
    acc[0] = __builtin_amdgcn_wmma_f32_16x16x32_bf16(false, a, false, b0,
                                                     (short)0, acc[0], false, false);
    acc[1] = __builtin_amdgcn_wmma_f32_16x16x32_bf16(false, a, false, b1,
                                                     (short)0, acc[1], false, false);
    acc[2] = __builtin_amdgcn_wmma_f32_16x16x32_bf16(false, a, false, b2,
                                                     (short)0, acc[2], false, false);
    acc[3] = __builtin_amdgcn_wmma_f32_16x16x32_bf16(false, a, false, b3,
                                                     (short)0, acc[3], false, false);
  }
  // C/D f32 16x16: VGPR r -> row r + 8*g, column = lane&15 (+16 per tile)
#pragma unroll
  for (int j = 0; j < 4; ++j) {
    int col = tc * 64 + 16 * j + mr;
#pragma unroll
    for (int r = 0; r < 8; ++r) {
      int orow = tm * 16 + 8 * g + r;
      if (orow < M) C[(size_t)orow * NC + col] = acc[j][r];
    }
  }
}

// ---------------- attention logits el/er: wave per (node, head) ----------------
__global__ void k_el_er(const float* __restrict__ ft, const float* __restrict__ al,
                        const float* __restrict__ ar, float* __restrict__ el,
                        float* __restrict__ er, int N) {
  int wave = (int)((blockIdx.x * blockDim.x + threadIdx.x) >> 5);
  int lane = threadIdx.x & 31;
  if (wave >= N * GAT_H) return;       // wave-uniform
  int n = wave >> 2, h = wave & 3;
  const float* f   = ft + (size_t)n * GAT_HD + h * GAT_D;
  const float* alh = al + h * GAT_D;
  const float* arh = ar + h * GAT_D;
  float f0 = f[lane], f1 = f[lane + 32];
  float sl = f0 * alh[lane] + f1 * alh[lane + 32];
  float sr = f0 * arh[lane] + f1 * arh[lane + 32];
#pragma unroll
  for (int off = 16; off >= 1; off >>= 1) {
    sl += __shfl_xor(sl, off, 32);
    sr += __shfl_xor(sr, off, 32);
  }
  if (lane == 0) { el[wave] = sl; er[wave] = sr; }
}

// ---------------- edge pass 1: leaky-relu logits + segment max ----------------
__global__ void k_edge_logits(const int* __restrict__ src, const int* __restrict__ dst,
                              const float* __restrict__ el, const float* __restrict__ er,
                              float* __restrict__ ebuf, float* __restrict__ m, int E) {
  int e = blockIdx.x * blockDim.x + threadIdx.x;
  if (e >= E) return;
  int s = src[e], d = dst[e];
  float4 l = ((const float4*)el)[s];
  float4 r = ((const float4*)er)[d];
  float4 x;
  x.x = l.x + r.x; x.y = l.y + r.y; x.z = l.z + r.z; x.w = l.w + r.w;
  x.x = x.x > 0.f ? x.x : NEG_SLOPE * x.x;
  x.y = x.y > 0.f ? x.y : NEG_SLOPE * x.y;
  x.z = x.z > 0.f ? x.z : NEG_SLOPE * x.z;
  x.w = x.w > 0.f ? x.w : NEG_SLOPE * x.w;
  ((float4*)ebuf)[e] = x;
  atomicMaxF(&m[d * 4 + 0], x.x);
  atomicMaxF(&m[d * 4 + 1], x.y);
  atomicMaxF(&m[d * 4 + 2], x.z);
  atomicMaxF(&m[d * 4 + 3], x.w);
}

// ---------------- edge pass 2: exp(e - m[dst]) + segment sum (in-place on ebuf) ----------------
__global__ void k_edge_exp(const int* __restrict__ dst, float* __restrict__ ebuf,
                           const float* __restrict__ m, float* __restrict__ den, int E) {
  int e = blockIdx.x * blockDim.x + threadIdx.x;
  if (e >= E) return;
  int d = dst[e];
  float4 x  = ((const float4*)ebuf)[e];
  float4 mm = ((const float4*)m)[d];
  float4 a;
  a.x = __expf(x.x - mm.x);
  a.y = __expf(x.y - mm.y);
  a.z = __expf(x.z - mm.z);
  a.w = __expf(x.w - mm.w);
  ((float4*)ebuf)[e] = a;
  atomicAdd(&den[d * 4 + 0], a.x);
  atomicAdd(&den[d * 4 + 1], a.y);
  atomicAdd(&den[d * 4 + 2], a.z);
  atomicAdd(&den[d * 4 + 3], a.w);
}

// ---------------- edge pass 3: wave per edge, scatter 256 f32 with atomics ----------------
__global__ void k_edge_scatter(const int* __restrict__ src, const int* __restrict__ dst,
                               const float* __restrict__ abuf, const float* __restrict__ den,
                               const float* __restrict__ ft, float* __restrict__ agg, int E) {
  int e = (int)((blockIdx.x * blockDim.x + threadIdx.x) >> 5);
  int lane = threadIdx.x & 31;
  if (e >= E) return;                  // wave-uniform
  int s = src[e], d = dst[e];
  int hl = lane & 3;
  float cl = abuf[(size_t)e * 4 + hl] / fmaxf(den[(size_t)d * 4 + hl], 1e-20f);
  float c0 = __shfl(cl, 0, 32), c1 = __shfl(cl, 1, 32);
  float c2 = __shfl(cl, 2, 32), c3 = __shfl(cl, 3, 32);
  const float* fr = ft  + (size_t)s * GAT_HD;
  float*       ar = agg + (size_t)d * GAT_HD;
#pragma unroll
  for (int i = 0; i < 8; ++i) {        // d-index = lane + 32*i; head = i>>1
    int dd = lane + 32 * i;
    float cf = (i < 2) ? c0 : (i < 4) ? c1 : (i < 6) ? c2 : c3;
    atomicAdd(&ar[dd], fr[dd] * cf);
  }
}

// ---------------- mean over heads + bias + LayerNorm + ReLU -> bf16 ----------------
__global__ void k_mean_ln_relu(const float* __restrict__ agg, const float* __restrict__ bias,
                               const float* __restrict__ gamma, const float* __restrict__ beta,
                               unsigned short* __restrict__ hb, int N) {
  int node = (int)((blockIdx.x * blockDim.x + threadIdx.x) >> 5);
  int lane = threadIdx.x & 31;
  if (node >= N) return;               // wave-uniform
  const float* a = agg + (size_t)node * GAT_HD;
  float x0 = 0.f, x1 = 0.f;
#pragma unroll
  for (int h = 0; h < GAT_H; ++h) {
    x0 += a[h * GAT_D + lane]      + bias[h * GAT_D + lane];
    x1 += a[h * GAT_D + lane + 32] + bias[h * GAT_D + lane + 32];
  }
  x0 *= 0.25f; x1 *= 0.25f;
  float s = x0 + x1;
#pragma unroll
  for (int off = 16; off >= 1; off >>= 1) s += __shfl_xor(s, off, 32);
  float mean = s * (1.0f / 64.0f);
  float d0 = x0 - mean, d1 = x1 - mean;
  float v = d0 * d0 + d1 * d1;
#pragma unroll
  for (int off = 16; off >= 1; off >>= 1) v += __shfl_xor(v, off, 32);
  float rstd = rsqrtf(v * (1.0f / 64.0f) + 1e-5f);
  float y0 = fmaxf(d0 * rstd * gamma[lane]      + beta[lane],      0.f);
  float y1 = fmaxf(d1 * rstd * gamma[lane + 32] + beta[lane + 32], 0.f);
  hb[(size_t)node * GAT_D + lane]      = f32_to_bf16_rne(y0);
  hb[(size_t)node * GAT_D + lane + 32] = f32_to_bf16_rne(y1);
}

// ---------------- final: mean over heads + bias -> d_out [N,64] f32 ----------------
__global__ void k_mean_bias_out(const float* __restrict__ agg, const float* __restrict__ bias,
                                float* __restrict__ out, int N) {
  int i = blockIdx.x * blockDim.x + threadIdx.x;
  if (i >= N * GAT_D) return;
  int n = i >> 6, d = i & 63;
  const float* a = agg + (size_t)n * GAT_HD;
  float s = 0.f;
#pragma unroll
  for (int h = 0; h < GAT_H; ++h) s += a[h * GAT_D + d] + bias[h * GAT_D + d];
  out[i] = 0.25f * s;
}

// ---------------- host-side orchestration ----------------
extern "C" void kernel_launch(void* const* d_in, const int* in_sizes, int n_in,
                              void* d_out, int out_size, void* d_ws, size_t ws_size,
                              hipStream_t stream) {
  const float* feat  = (const float*)d_in[0];
  const int*   src   = (const int*)  d_in[1];
  const int*   dst   = (const int*)  d_in[2];
  const float* W1    = (const float*)d_in[3];
  const float* al1   = (const float*)d_in[4];
  const float* ar1   = (const float*)d_in[5];
  const float* b1    = (const float*)d_in[6];
  const float* W2    = (const float*)d_in[7];
  const float* al2   = (const float*)d_in[8];
  const float* ar2   = (const float*)d_in[9];
  const float* b2    = (const float*)d_in[10];
  const float* gamma = (const float*)d_in[11];
  const float* beta  = (const float*)d_in[12];

  const int N = in_sizes[0] / GAT_HD;   // 50000
  const int E = in_sizes[1];            // 800000
  const int DIN = GAT_HD;               // 256

  // workspace carve-up (256B aligned slices)
  char* p = (char*)d_ws;
  auto take = [&](size_t bytes) -> void* {
    void* r = (void*)p;
    p += (bytes + 255) & ~(size_t)255;
    return r;
  };
  unsigned short* featb = (unsigned short*)take((size_t)N * DIN * 2);
  unsigned short* W1t   = (unsigned short*)take((size_t)DIN * GAT_HD * 2);   // [NC][K]
  unsigned short* W2t   = (unsigned short*)take((size_t)GAT_D * GAT_HD * 2); // [NC][K]
  unsigned short* hb    = (unsigned short*)take((size_t)N * GAT_D * 2);
  float* ft   = (float*)take((size_t)N * GAT_HD * 4);   // GEMM out, reused both layers
  float* agg  = (float*)take((size_t)N * GAT_HD * 4);   // scatter target, reused
  float* el   = (float*)take((size_t)N * GAT_H * 4);
  float* er   = (float*)take((size_t)N * GAT_H * 4);
  float* m    = (float*)take((size_t)N * GAT_H * 4);
  float* den  = (float*)take((size_t)N * GAT_H * 4);
  float* ebuf = (float*)take((size_t)E * GAT_H * 4);    // logits, then exp'd (in place)

  const int B256 = 256;
  auto gridT = [&](int count) { return (count + B256 - 1) / B256; };
  auto gridW = [&](int waves) { return (waves + 7) / 8; };  // 8 waves / 256-thread block

  // --- one-time conversions to bf16 (weights transposed for contiguous B frags) ---
  k_cvt_bf16<<<gridT(N * DIN), B256, 0, stream>>>(feat, featb, N * DIN);
  k_cvt_bf16_t<<<gridT(DIN * GAT_HD), B256, 0, stream>>>(W1, W1t, DIN, GAT_HD);
  k_cvt_bf16_t<<<gridT(GAT_D * GAT_HD), B256, 0, stream>>>(W2, W2t, GAT_D, GAT_HD);

  const int stripsN = GAT_HD / 64;
  const int tilesM  = (N + 15) / 16;
  const int gemmWaves  = tilesM * stripsN;
  const int gemmBlocks = (gemmWaves + 3) / 4;  // 128-thread blocks = 4 waves

  // ================= layer 1 =================
  k_gemm_wmma_bf16<256, 256><<<gemmBlocks, 128, 0, stream>>>(featb, W1t, ft, N);
  k_el_er<<<gridW(N * GAT_H), B256, 0, stream>>>(ft, al1, ar1, el, er, N);
  k_fill<<<gridT(N * GAT_H), B256, 0, stream>>>(m, -INFINITY, N * GAT_H);
  k_fill<<<gridT(N * GAT_H), B256, 0, stream>>>(den, 0.f, N * GAT_H);
  k_fill<<<gridT(N * GAT_HD), B256, 0, stream>>>(agg, 0.f, N * GAT_HD);
  k_edge_logits<<<gridT(E), B256, 0, stream>>>(src, dst, el, er, ebuf, m, E);
  k_edge_exp<<<gridT(E), B256, 0, stream>>>(dst, ebuf, m, den, E);
  k_edge_scatter<<<gridW(E), B256, 0, stream>>>(src, dst, ebuf, den, ft, agg, E);
  k_mean_ln_relu<<<gridW(N), B256, 0, stream>>>(agg, b1, gamma, beta, hb, N);

  // ================= layer 2 =================
  k_gemm_wmma_bf16<64, 256><<<gemmBlocks, 128, 0, stream>>>(hb, W2t, ft, N);
  k_el_er<<<gridW(N * GAT_H), B256, 0, stream>>>(ft, al2, ar2, el, er, N);
  k_fill<<<gridT(N * GAT_H), B256, 0, stream>>>(m, -INFINITY, N * GAT_H);
  k_fill<<<gridT(N * GAT_H), B256, 0, stream>>>(den, 0.f, N * GAT_H);
  k_fill<<<gridT(N * GAT_HD), B256, 0, stream>>>(agg, 0.f, N * GAT_HD);
  k_edge_logits<<<gridT(E), B256, 0, stream>>>(src, dst, el, er, ebuf, m, E);
  k_edge_exp<<<gridT(E), B256, 0, stream>>>(dst, ebuf, m, den, E);
  k_edge_scatter<<<gridW(E), B256, 0, stream>>>(src, dst, ebuf, den, ft, agg, E);
  k_mean_bias_out<<<gridT(N * GAT_D), B256, 0, stream>>>(agg, b2, (float*)d_out, N);
}
